// VisionMamba_69621419868438
// MI455X (gfx1250) — matrix operand order repaired and checked by reading, not compiled, not run on hardware
//
#include <hip/hip_runtime.h>
#include <hip/hip_bf16.h>

// ---------------- model constants ----------------
#define DEPTH   24
#define DMODEL  384
#define DINNER  768
#define DSTATE  16
#define DTRANK  24
#define DCONV   3
#define PATCH   16
#define IMG     224
#define NPATCH  196          // (224/16)^2
#define BATCH   4
#define SEQL    392          // 2 * NPATCH
#define BL      (BATCH*SEQL) // 1568 tokens
#define NXPAD   64           // x_proj output cols padded 56 -> 64

typedef __bf16 bf16_t;
typedef bf16_t v16bf __attribute__((ext_vector_type(16)));
typedef bf16_t v8bf  __attribute__((ext_vector_type(8)));
typedef float  v8f   __attribute__((ext_vector_type(8)));

union BF16Frag { v16bf v; v8bf h[2]; };

// ---------------- weight conversion ----------------
__global__ void f32_to_bf16_kernel(const float* __restrict__ src,
                                   bf16_t* __restrict__ dst, size_t n) {
    size_t i = (size_t)blockIdx.x * blockDim.x + threadIdx.x;
    if (i < n) dst[i] = (bf16_t)src[i];
}

// pad x_proj weights (DEPTH,56,768) -> (DEPTH,64,768) bf16, zero rows 56..63
__global__ void xproj_pad_kernel(const float* __restrict__ src,
                                 bf16_t* __restrict__ dst) {
    size_t i = (size_t)blockIdx.x * blockDim.x + threadIdx.x;
    const size_t total = (size_t)DEPTH * NXPAD * DINNER;
    if (i >= total) return;
    size_t l   = i / ((size_t)NXPAD * DINNER);
    size_t rem = i % ((size_t)NXPAD * DINNER);
    size_t row = rem / DINNER;
    size_t k   = rem % DINNER;
    dst[i] = (row < (DTRANK + 2 * DSTATE))
               ? (bf16_t)src[(l * (DTRANK + 2 * DSTATE) + row) * DINNER + k]
               : (bf16_t)0.0f;
}

// ---------------- patch embedding (one-shot, fp32 VALU) ----------------
__global__ void patch_embed_kernel(const float* __restrict__ z,
                                   const float* __restrict__ x,
                                   const float* __restrict__ pw,
                                   const float* __restrict__ pb,
                                   const float* __restrict__ pos,
                                   float* __restrict__ h) {
    __shared__ float patch[768];
    int t = blockIdx.x;          // token 0..1567
    int b = t / SEQL;
    int l = t % SEQL;
    int s = l / NPATCH;          // 0 = z stream, 1 = x stream
    int p = l % NPATCH;
    int py = p / (IMG / PATCH), px = p % (IMG / PATCH);
    const float* img = (s == 0 ? z : x) + (size_t)b * 3 * IMG * IMG;
    for (int i = threadIdx.x; i < 768; i += blockDim.x) {
        int ci = i >> 8, rem = i & 255;
        int ii = rem >> 4, jj = rem & 15;
        patch[i] = img[(size_t)ci * IMG * IMG + (size_t)(py * PATCH + ii) * IMG
                       + (px * PATCH + jj)];
    }
    __syncthreads();
    for (int c = threadIdx.x; c < DMODEL; c += blockDim.x) {
        const float* wrow = pw + (size_t)c * 768;
        float acc = pb[c] + pos[(size_t)(p + 1) * DMODEL + c];
        for (int k = 0; k < 768; ++k) acc += patch[k] * wrow[k];
        h[(size_t)t * DMODEL + c] = acc;
    }
}

// ---------------- layernorm over 384 cols (block = 128 threads) -------------
__global__ void ln_kernel(const float* __restrict__ X,
                          const float* __restrict__ w,
                          const float* __restrict__ b,
                          bf16_t* __restrict__ out_bf,
                          float* __restrict__ out_f) {
    __shared__ float s1[128], s2[128];
    int t = blockIdx.x;
    const float* xr = X + (size_t)t * DMODEL;
    int tid = threadIdx.x;
    float v0 = xr[tid], v1 = xr[tid + 128], v2 = xr[tid + 256];
    s1[tid] = v0 + v1 + v2;
    s2[tid] = v0 * v0 + v1 * v1 + v2 * v2;
    __syncthreads();
    for (int o = 64; o > 0; o >>= 1) {
        if (tid < o) { s1[tid] += s1[tid + o]; s2[tid] += s2[tid + o]; }
        __syncthreads();
    }
    float mu  = s1[0] * (1.0f / DMODEL);
    float var = s2[0] * (1.0f / DMODEL) - mu * mu;
    float rs  = rsqrtf(var + 1e-5f);
    #pragma unroll
    for (int j = 0; j < 3; ++j) {
        int c = tid + j * 128;
        float v = (j == 0 ? v0 : (j == 1 ? v1 : v2));
        float y = (v - mu) * rs * w[c] + b[c];
        if (out_bf) out_bf[(size_t)t * DMODEL + c] = (bf16_t)y;
        if (out_f)  out_f[(size_t)t * DMODEL + c]  = y;
    }
}

// ---------------- WMMA fragment loads (ISA VGPR layouts) ----------------
__device__ __forceinline__ v16bf load_fragA(const bf16_t* __restrict__ base,
                                            int row, int ld, int k0, int lane) {
    // lanes 0-15: K = {k0..k0+7, k0+16..k0+23}; lanes 16-31: +8
    int half = lane >> 4, r = lane & 15;
    const bf16_t* p = base + (size_t)(row + r) * ld + k0 + half * 8;
    BF16Frag f;
    f.h[0] = *(const v8bf*)p;
    f.h[1] = *(const v8bf*)(p + 16);
    return f.v;
}
__device__ __forceinline__ v16bf load_fragB(const bf16_t* __restrict__ base,
                                            int col, int ld, int k0, int lane) {
    // lane%16 = column; lanes 0-15: K = {k0..k0+7, k0+16..k0+23}; lanes 16-31: +8
    int half = lane >> 4, n = lane & 15;
    const bf16_t* p = base + (size_t)(col + n) * ld + k0 + half * 8;
    BF16Frag f;
    f.h[0] = *(const v8bf*)p;
    f.h[1] = *(const v8bf*)(p + 16);
    return f.v;
}

#define WMMA_BF16(A_, B_, C_) \
    __builtin_amdgcn_wmma_f32_16x16x32_bf16(false, (A_), false, (B_), (short)0, (C_), false, false)

// ---------------- simple GEMM: one 16x16 tile per wave (small-N path) -------
__global__ void gemm_bf16_16x16_kernel(const bf16_t* __restrict__ A,
                                       const bf16_t* __restrict__ Bt,
                                       float* __restrict__ C,
                                       int M, int N, int K, int accumulate) {
    int wave = threadIdx.x >> 5;
    int lane = threadIdx.x & 31;
    int tile = blockIdx.x * (blockDim.x >> 5) + wave;
    int ntN = N >> 4;
    int ntTot = (M >> 4) * ntN;
    if (tile >= ntTot) return;
    int m0 = (tile / ntN) << 4;
    int n0 = (tile % ntN) << 4;
    v8f acc = {0.f, 0.f, 0.f, 0.f, 0.f, 0.f, 0.f, 0.f};
    for (int k0 = 0; k0 < K; k0 += 32) {
        v16bf a = load_fragA(A, m0, K, k0, lane);
        v16bf b = load_fragB(Bt, n0, K, k0, lane);
        acc = WMMA_BF16(a, b, acc);
    }
    int half = lane >> 4, col = lane & 15;
    float* cp = C + (size_t)(m0 + half * 8) * N + n0 + col;
    if (accumulate) {
        #pragma unroll
        for (int r = 0; r < 8; ++r) cp[(size_t)r * N] += acc[r];
    } else {
        #pragma unroll
        for (int r = 0; r < 8; ++r) cp[(size_t)r * N] = acc[r];
    }
}

// ---------------- register-blocked GEMM: 32(M)x64(N) tile per wave ----------
// Per K=32 step: 2 A-frags + 4 B-frags (12 x b128) feed 8 WMMAs
// (1.5 loads per v_wmma vs 4 in the naive kernel).
__global__ void gemm_bf16_32x64_kernel(const bf16_t* __restrict__ A,
                                       const bf16_t* __restrict__ Bt,
                                       float* __restrict__ C,
                                       int M, int N, int K, int accumulate) {
    int wave = threadIdx.x >> 5;
    int lane = threadIdx.x & 31;
    int tile = blockIdx.x * (blockDim.x >> 5) + wave;
    int ntN = N >> 6;                 // 64-wide N tiles
    int ntTot = (M >> 5) * ntN;       // 32-tall M tiles
    if (tile >= ntTot) return;
    int m0 = (tile / ntN) << 5;
    int n0 = (tile % ntN) << 6;

    v8f acc[2][4];
    #pragma unroll
    for (int i = 0; i < 2; ++i)
        #pragma unroll
        for (int j = 0; j < 4; ++j)
            acc[i][j] = (v8f){0.f, 0.f, 0.f, 0.f, 0.f, 0.f, 0.f, 0.f};

    for (int k0 = 0; k0 < K; k0 += 32) {
        v16bf a0 = load_fragA(A, m0,      K, k0, lane);
        v16bf a1 = load_fragA(A, m0 + 16, K, k0, lane);
        v16bf b0 = load_fragB(Bt, n0,      K, k0, lane);
        v16bf b1 = load_fragB(Bt, n0 + 16, K, k0, lane);
        v16bf b2 = load_fragB(Bt, n0 + 32, K, k0, lane);
        v16bf b3 = load_fragB(Bt, n0 + 48, K, k0, lane);
        acc[0][0] = WMMA_BF16(a0, b0, acc[0][0]);
        acc[0][1] = WMMA_BF16(a0, b1, acc[0][1]);
        acc[0][2] = WMMA_BF16(a0, b2, acc[0][2]);
        acc[0][3] = WMMA_BF16(a0, b3, acc[0][3]);
        acc[1][0] = WMMA_BF16(a1, b0, acc[1][0]);
        acc[1][1] = WMMA_BF16(a1, b1, acc[1][1]);
        acc[1][2] = WMMA_BF16(a1, b2, acc[1][2]);
        acc[1][3] = WMMA_BF16(a1, b3, acc[1][3]);
    }

    int half = lane >> 4, col = lane & 15;
    #pragma unroll
    for (int i = 0; i < 2; ++i) {
        #pragma unroll
        for (int j = 0; j < 4; ++j) {
            float* cp = C + (size_t)(m0 + i * 16 + half * 8) * N + n0 + j * 16 + col;
            if (accumulate) {
                #pragma unroll
                for (int r = 0; r < 8; ++r) cp[(size_t)r * N] += acc[i][j][r];
            } else {
                #pragma unroll
                for (int r = 0; r < 8; ++r) cp[(size_t)r * N] = acc[i][j][r];
            }
        }
    }
}

// ---------------- depthwise causal conv(3) + SiLU ----------------
__global__ void conv_silu_kernel(const float* __restrict__ xz,  // [BL,1536]
                                 const float* __restrict__ cw,  // [768,3]
                                 const float* __restrict__ cb,  // [768]
                                 float* __restrict__ xc_f,
                                 bf16_t* __restrict__ xc_bf) {
    size_t i = (size_t)blockIdx.x * blockDim.x + threadIdx.x;
    if (i >= (size_t)BL * DINNER) return;
    int t = (int)(i / DINNER), e = (int)(i % DINNER);
    int l = t % SEQL;
    float w0 = cw[e * 3 + 0], w1 = cw[e * 3 + 1], w2 = cw[e * 3 + 2];
    float v = cb[e];
    if (l >= 2) v += xz[(size_t)(t - 2) * (2 * DINNER) + e] * w0;
    if (l >= 1) v += xz[(size_t)(t - 1) * (2 * DINNER) + e] * w1;
    v += xz[(size_t)t * (2 * DINNER) + e] * w2;
    float s = v / (1.0f + __expf(-v));   // SiLU
    xc_f[i]  = s;
    xc_bf[i] = (bf16_t)s;
}

// ---------------- dt = softplus(dtout @ dwT + db) ----------------
__global__ void dt_kernel(const float* __restrict__ xpo,  // [BL,64], cols 0..23 = dt rank
                          const float* __restrict__ dw,   // [768,24]
                          const float* __restrict__ db,   // [768]
                          float* __restrict__ dt) {
    size_t i = (size_t)blockIdx.x * blockDim.x + threadIdx.x;
    if (i >= (size_t)BL * DINNER) return;
    int t = (int)(i / DINNER), d = (int)(i % DINNER);
    const float* xr = xpo + (size_t)t * NXPAD;
    const float* wr = dw + (size_t)d * DTRANK;
    float acc = db[d];
    #pragma unroll
    for (int r = 0; r < DTRANK; ++r) acc += xr[r] * wr[r];
    dt[i] = (acc > 20.0f) ? acc : log1pf(__expf(acc));   // stable softplus
}

// ---------------- selective scan: 16 lanes per (b,d), shfl_xor reduce -------
__global__ void scan_kernel(const float* __restrict__ dt,   // [BL,768]
                            const float* __restrict__ xc,   // [BL,768]
                            const float* __restrict__ xpo,  // [BL,64]: B=24..39, C=40..55
                            const float* __restrict__ xz,   // [BL,1536]: zg = cols 768..
                            const float* __restrict__ A_log,// [768,16]
                            const float* __restrict__ Dp,   // [768]
                            bf16_t* __restrict__ ybf) {     // [BL,768]
    int gid = blockIdx.x * blockDim.x + threadIdx.x;
    int pair = gid >> 4;
    int n = gid & 15;
    if (pair >= BATCH * DINNER) return;
    int b = pair / DINNER, d = pair % DINNER;
    float An = -__expf(A_log[d * DSTATE + n]);
    float dd = Dp[d];
    float h = 0.0f;
    for (int l = 0; l < SEQL; ++l) {
        int t = b * SEQL + l;
        float dtv = dt[(size_t)t * DINNER + d];
        float xcv = xc[(size_t)t * DINNER + d];
        float Bn  = xpo[(size_t)t * NXPAD + DTRANK + n];
        float Cn  = xpo[(size_t)t * NXPAD + DTRANK + DSTATE + n];
        h = __expf(dtv * An) * h + dtv * xcv * Bn;
        float ps = h * Cn;
        ps += __shfl_xor(ps, 1);
        ps += __shfl_xor(ps, 2);
        ps += __shfl_xor(ps, 4);
        ps += __shfl_xor(ps, 8);
        if (n == 0) {
            float y  = ps + dd * xcv;
            float zg = xz[(size_t)t * (2 * DINNER) + DINNER + d];
            float sg = zg / (1.0f + __expf(-zg));   // SiLU gate
            ybf[(size_t)t * DINNER + d] = (bf16_t)(y * sg);
        }
    }
}

// =========================== host side =============================
extern "C" void kernel_launch(void* const* d_in, const int* in_sizes, int n_in,
                              void* d_out, int out_size, void* d_ws, size_t ws_size,
                              hipStream_t stream) {
    const float* z        = (const float*)d_in[0];
    const float* x        = (const float*)d_in[1];
    const float* patch_W  = (const float*)d_in[2];
    const float* patch_b  = (const float*)d_in[3];
    const float* pos_emb  = (const float*)d_in[4];
    const float* norm_w   = (const float*)d_in[5];
    const float* norm_b   = (const float*)d_in[6];
    const float* in_W     = (const float*)d_in[7];
    const float* conv_W   = (const float*)d_in[8];
    const float* conv_b   = (const float*)d_in[9];
    const float* xproj_W  = (const float*)d_in[10];
    const float* dtproj_W = (const float*)d_in[11];
    const float* dtproj_b = (const float*)d_in[12];
    const float* A_log    = (const float*)d_in[13];
    const float* D_param  = (const float*)d_in[14];
    const float* out_W    = (const float*)d_in[15];
    const float* fnorm_w  = (const float*)d_in[16];
    const float* fnorm_b  = (const float*)d_in[17];
    float* out = (float*)d_out;

    // ---- workspace partition ----
    char* ws = (char*)d_ws;
    size_t off = 0;
    auto take = [&](size_t bytes) -> char* {
        char* p = ws + off;
        off += (bytes + 255) & ~(size_t)255;
        return p;
    };
    bf16_t* wb_in  = (bf16_t*)take((size_t)DEPTH * 2 * DINNER * DMODEL * sizeof(bf16_t));
    bf16_t* wb_xp  = (bf16_t*)take((size_t)DEPTH * NXPAD * DINNER * sizeof(bf16_t));
    bf16_t* wb_out = (bf16_t*)take((size_t)DEPTH * DMODEL * DINNER * sizeof(bf16_t));
    float*  hbuf   = (float*) take((size_t)BL * DMODEL * sizeof(float));
    bf16_t* xln    = (bf16_t*)take((size_t)BL * DMODEL * sizeof(bf16_t));
    float*  xz     = (float*) take((size_t)BL * 2 * DINNER * sizeof(float));
    float*  xc_f   = (float*) take((size_t)BL * DINNER * sizeof(float));
    bf16_t* xc_bf  = (bf16_t*)take((size_t)BL * DINNER * sizeof(bf16_t));
    float*  xpo    = (float*) take((size_t)BL * NXPAD * sizeof(float));
    float*  dtb    = (float*) take((size_t)BL * DINNER * sizeof(float));
    bf16_t* ybf    = (bf16_t*)take((size_t)BL * DINNER * sizeof(bf16_t));
    (void)ws_size; (void)in_sizes; (void)n_in; (void)out_size;

    // ---- one-shot: weight conversion to bf16 ----
    {
        size_t n_in_w  = (size_t)DEPTH * 2 * DINNER * DMODEL;
        size_t n_out_w = (size_t)DEPTH * DMODEL * DINNER;
        size_t n_xp    = (size_t)DEPTH * NXPAD * DINNER;
        f32_to_bf16_kernel<<<(unsigned)((n_in_w + 255) / 256), 256, 0, stream>>>(in_W, wb_in, n_in_w);
        f32_to_bf16_kernel<<<(unsigned)((n_out_w + 255) / 256), 256, 0, stream>>>(out_W, wb_out, n_out_w);
        xproj_pad_kernel<<<(unsigned)((n_xp + 255) / 256), 256, 0, stream>>>(xproj_W, wb_xp);
    }

    // ---- one-shot: patch embedding -> hbuf ----
    patch_embed_kernel<<<BL, 256, 0, stream>>>(z, x, patch_W, patch_b, pos_emb, hbuf);

    // ---- per-layer pipeline ----
    const int elemGrid = (BL * DINNER + 255) / 256;
    for (int l = 0; l < DEPTH; ++l) {
        // 1. layernorm -> bf16
        ln_kernel<<<BL, 128, 0, stream>>>(hbuf, norm_w + (size_t)l * DMODEL,
                                          norm_b + (size_t)l * DMODEL, xln, nullptr);
        // 2. in_proj: [BL,384] x [1536,384]^T -> [BL,1536]   (32x64 blocked)
        {
            int tiles = (BL / 32) * (2 * DINNER / 64);   // 49 * 24
            gemm_bf16_32x64_kernel<<<(tiles + 7) / 8, 256, 0, stream>>>(
                xln, wb_in + (size_t)l * 2 * DINNER * DMODEL, xz,
                BL, 2 * DINNER, DMODEL, 0);
        }
        // 3. depthwise causal conv + SiLU
        conv_silu_kernel<<<elemGrid, 256, 0, stream>>>(
            xz, conv_W + (size_t)l * DINNER * DCONV, conv_b + (size_t)l * DINNER,
            xc_f, xc_bf);
        // 4. x_proj: [BL,768] x [64,768]^T -> [BL,64]  (N padded 56->64)
        //    small N -> 16x16-per-wave kernel keeps 392 waves of parallelism
        {
            int tiles = (BL / 16) * (NXPAD / 16);        // 98 * 4
            gemm_bf16_16x16_kernel<<<(tiles + 7) / 8, 256, 0, stream>>>(
                xc_bf, wb_xp + (size_t)l * NXPAD * DINNER, xpo,
                BL, NXPAD, DINNER, 0);
        }
        // 5. dt projection + softplus
        dt_kernel<<<elemGrid, 256, 0, stream>>>(
            xpo, dtproj_W + (size_t)l * DINNER * DTRANK,
            dtproj_b + (size_t)l * DINNER, dtb);
        // 6. selective scan (fused D-skip + SiLU gate), emits bf16
        {
            int threads = BATCH * DINNER * DSTATE;   // 49152
            scan_kernel<<<(threads + 255) / 256, 256, 0, stream>>>(
                dtb, xc_f, xpo, xz, A_log + (size_t)l * DINNER * DSTATE,
                D_param + (size_t)l * DINNER, ybf);
        }
        // 7. out_proj + residual: hbuf += [BL,768] x [384,768]^T  (32x64 blocked)
        {
            int tiles = (BL / 32) * (DMODEL / 64);       // 49 * 6
            gemm_bf16_32x64_kernel<<<(tiles + 7) / 8, 256, 0, stream>>>(
                ybf, wb_out + (size_t)l * DMODEL * DINNER, hbuf,
                BL, DMODEL, DINNER, 1);
        }
    }

    // ---- final layernorm -> d_out (fp32) ----
    ln_kernel<<<BL, 128, 0, stream>>>(hbuf, fnorm_w, fnorm_b, nullptr, out);
}
